// DecoderAttModule_41515153883139
// MI455X (gfx1250) — matrix operand order repaired and checked by reading, not compiled, not run on hardware
//
#include <hip/hip_runtime.h>
#include <hip/hip_bf16.h>
#include <math.h>

// ---------------- problem dims ----------------
#define B_ 64
#define R_ 36
#define F_ 2048
#define E_ 1024
#define D_ 1024
#define A_ 512
#define V_ 32000
#define L_ 21
#define T_ 20   // MAX_DEC

typedef __bf16 bf16_t;
typedef __attribute__((ext_vector_type(16))) __bf16 v16bf;
typedef __attribute__((ext_vector_type(8)))  __bf16 v8bf;
typedef __attribute__((ext_vector_type(8)))  float  v8f;

// K-chunk staged in LDS (all K dims are multiples of 128)
#define KC   128
#define KCP  (KC + 8)           // +16B row pad -> banks rotate by 4 per row
#define LDS_ELEMS (2 * 64 * KCP)

// ---- async global->LDS path (gate on toolchain support) ----
#if defined(__HIP_DEVICE_COMPILE__) && \
    __has_builtin(__builtin_amdgcn_global_load_async_to_lds_b128) && \
    __has_builtin(__builtin_amdgcn_s_wait_asynccnt)
#define USE_ASYNC_LDS 1
typedef int v4i_t __attribute__((ext_vector_type(4)));
typedef __attribute__((address_space(1))) v4i_t* as1_v4i;
typedef __attribute__((address_space(3))) v4i_t* as3_v4i;
#else
#define USE_ASYNC_LDS 0
#endif

// ================= WMMA fragment helpers =================
// B fragment: 32x16 bf16 of W^T, W (N,K) row-major: col n = lane&15,
// K base = (lane>>4)*16, 16 consecutive K per lane (same K order as the A fragment).
__device__ __forceinline__ v16bf load_b_frag(const bf16_t* __restrict__ W,
                                             int ldw, int n0, int k0, int lane) {
    const bf16_t* p = W + (size_t)(n0 + (lane & 15)) * ldw + k0 + ((lane >> 4) << 4);
    __builtin_prefetch((const void*)(p + 128), 0, 1);  // global_prefetch_b8
    return *(const v16bf*)(p);
}

// A fragment: 16x32 bf16 from LDS strip: row m = lane&15, K base = (lane>>4)*16,
// VGPR0..7 = K..K+15 (matches B fragment).
__device__ __forceinline__ v16bf load_a_frag_lds(const bf16_t* buf, int m0,
                                                 int kk, int lane) {
    const bf16_t* p = buf + (m0 + (lane & 15)) * KCP + kk + ((lane >> 4) << 4);
    v8bf lo = *(const v8bf*)(p);
    v8bf hi = *(const v8bf*)(p + 8);
    v16bf r;
#pragma unroll
    for (int i = 0; i < 8; ++i) { r[i] = lo[i]; r[i + 8] = hi[i]; }
    return r;
}

__device__ __forceinline__ v8f wmma_bf16(v16bf a, v16bf b, v8f c) {
    return __builtin_amdgcn_wmma_f32_16x16x32_bf16(
        false, a, false, b, (short)0, c, false, false);
}

// Stage one 64 x KC bf16 chunk of A into an LDS buffer (256 threads).
// 1024 16-byte units; lanes copy consecutive units for coalescing.
__device__ __forceinline__ void stage_chunk(const bf16_t* __restrict__ A, int lda,
                                            int m0, int k0, bf16_t* buf, int tid) {
#pragma unroll
    for (int i = 0; i < 4; ++i) {
        const int u = i * 256 + tid;
        const int row = u >> 4;          // 16 units per 128-elem row
        const int unit = u & 15;
        const bf16_t* g = A + (size_t)(m0 + row) * lda + k0 + unit * 8;
        bf16_t* l = buf + row * KCP + unit * 8;
#if USE_ASYNC_LDS
        __builtin_amdgcn_global_load_async_to_lds_b128(
            (as1_v4i)(void*)g, (as3_v4i)(void*)l, 0, 0);
#else
        *(v8bf*)l = *(const v8bf*)g;
#endif
    }
}

__device__ __forceinline__ void wait_stage() {
#if USE_ASYNC_LDS
    __builtin_amdgcn_s_wait_asynccnt(0);
#endif
    __syncthreads();
}

// One GEMM segment: acc[0..3] += A(m0..m0+63, 0..K) * W(n0..n0+15, 0..K)^T.
// All 8 waves call with identical A/K (barriers inside). K % KC == 0.
__device__ __forceinline__ void gemm_segment(const bf16_t* __restrict__ A, int lda,
                                             int m0,
                                             const bf16_t* __restrict__ W, int ldw,
                                             int n0, int K,
                                             v8f acc[4], bf16_t* lds,
                                             int lane, int tid) {
    const int nchunks = K / KC;
    stage_chunk(A, lda, m0, 0, lds, tid);
    for (int c = 0; c < nchunks; ++c) {
        bf16_t* cur = lds + (c & 1) * (64 * KCP);
        wait_stage();  // chunk c resident for every wave
        if (c + 1 < nchunks)
            stage_chunk(A, lda, m0, (c + 1) * KC, lds + ((c + 1) & 1) * (64 * KCP), tid);
#pragma unroll
        for (int kk = 0; kk < KC; kk += 32) {
            v16bf b = load_b_frag(W, ldw, n0, c * KC + kk, lane);
#pragma unroll
            for (int mt = 0; mt < 4; ++mt) {
                v16bf a = load_a_frag_lds(cur, mt << 4, kk, lane);
                acc[mt] = wmma_bf16(a, b, acc[mt]);
            }
        }
        __syncthreads();  // done reading buf before it is restaged
    }
}

// ================= GEMM kernels =================
// block = (32,8): 8 waves, each wave owns a 64x16 output strip (4 WMMA tiles).
// grid.x = N/128 (exact), grid.y = M/64 (exact) -> no inactive waves.

__global__ void gemm_kernel(const bf16_t* __restrict__ A, int lda,
                            const bf16_t* __restrict__ W, int ldw, int K,
                            const float* __restrict__ bias,
                            float* __restrict__ Dst, int N) {
    __shared__ bf16_t lds[LDS_ELEMS];
    const int lane = threadIdx.x;
    const int tid = threadIdx.y * 32 + lane;
    const int n0 = (blockIdx.x * 8 + threadIdx.y) << 4;
    const int m0 = blockIdx.y * 64;
    v8f acc[4] = {};
    gemm_segment(A, lda, m0, W, ldw, n0, K, acc, lds, lane, tid);
    const int n = n0 + (lane & 15);
    const float bs = bias ? bias[n] : 0.f;
#pragma unroll
    for (int mt = 0; mt < 4; ++mt) {
        const int mb = m0 + (mt << 4) + ((lane >> 4) << 3);
#pragma unroll
        for (int v = 0; v < 8; ++v)
            Dst[(size_t)(mb + v) * N + n] = acc[mt][v] + bs;
    }
}

// D(64,N) = A1*W1^T + A2*W2^T + b0 + b1 (fused LSTM input+hidden GEMM)
__global__ void gemm_dual_kernel(const bf16_t* __restrict__ A1, int lda1, int K1,
                                 const bf16_t* __restrict__ W1, int ldw1,
                                 const bf16_t* __restrict__ A2, int lda2, int K2,
                                 const bf16_t* __restrict__ W2, int ldw2,
                                 const float* __restrict__ b0,
                                 const float* __restrict__ b1,
                                 float* __restrict__ Dst, int N) {
    __shared__ bf16_t lds[LDS_ELEMS];
    const int lane = threadIdx.x;
    const int tid = threadIdx.y * 32 + lane;
    const int n0 = (blockIdx.x * 8 + threadIdx.y) << 4;
    v8f acc[4] = {};
    gemm_segment(A1, lda1, 0, W1, ldw1, n0, K1, acc, lds, lane, tid);
    gemm_segment(A2, lda2, 0, W2, ldw2, n0, K2, acc, lds, lane, tid);
    const int n = n0 + (lane & 15);
    const float bs = b0[n] + b1[n];
#pragma unroll
    for (int mt = 0; mt < 4; ++mt) {
        const int mb = (mt << 4) + ((lane >> 4) << 3);
#pragma unroll
        for (int v = 0; v < 8; ++v)
            Dst[(size_t)(mb + v) * N + n] = acc[mt][v] + bs;
    }
}

// preds[m, t, :] = mask ? nh2*Wo^T + bo : 0   (dominant GEMM, 64x32000x1024)
__global__ void logits_kernel(const bf16_t* __restrict__ A,
                              const bf16_t* __restrict__ Wo,
                              const float* __restrict__ bo,
                              const int* __restrict__ dec_len, int t,
                              float* __restrict__ preds) {
    __shared__ bf16_t lds[LDS_ELEMS];
    const int lane = threadIdx.x;
    const int tid = threadIdx.y * 32 + lane;
    const int n0 = (blockIdx.x * 8 + threadIdx.y) << 4;
    v8f acc[4] = {};
    gemm_segment(A, D_, 0, Wo, D_, n0, D_, acc, lds, lane, tid);
    const int n = n0 + (lane & 15);
    const float bs = bo[n];
#pragma unroll
    for (int mt = 0; mt < 4; ++mt) {
        const int mb = (mt << 4) + ((lane >> 4) << 3);
#pragma unroll
        for (int v = 0; v < 8; ++v) {
            const int m = mb + v;
            const bool msk = t < dec_len[m];
            preds[((size_t)m * T_ + t) * V_ + n] = msk ? (acc[mt][v] + bs) : 0.f;
        }
    }
}

// ================= prep / pointwise kernels =================
__global__ void cvt_f32_bf16_kernel(const float* __restrict__ s,
                                    bf16_t* __restrict__ d, int n) {
    int i = blockIdx.x * blockDim.x + threadIdx.x;
    if (i < n) d[i] = (bf16_t)s[i];
}

// stable argsort of -sizes on one block of 64 threads; also emits the int
// outputs (as f32) and the sorted token buffer.
__global__ void sort_kernel(const int* __restrict__ sizes,
                            const int* __restrict__ sequences,
                            int* __restrict__ pos, int* __restrict__ dec_len,
                            int* __restrict__ seqs_s,
                            float* __restrict__ out_seqs,
                            float* __restrict__ out_dlen,
                            float* __restrict__ out_pos) {
    __shared__ int s[B_];
    const int b = threadIdx.x;
    if (b >= B_) return;
    s[b] = sizes[b];
    __syncthreads();
    const int my = s[b];
    int rank = 0;
    for (int j = 0; j < B_; ++j) {
        const int sj = s[j];
        if (sj > my || (sj == my && j < b)) rank++;
    }
    pos[rank] = b;
    __syncthreads();
    const int src = pos[b];
    const int dl = s[src] - 1;
    dec_len[b] = dl;
    out_dlen[b] = (float)dl;
    out_pos[b] = (float)src;
    for (int l = 0; l < L_; ++l) {
        const int tok = sequences[src * L_ + l];
        seqs_s[b * L_ + l] = tok;
        out_seqs[b * L_ + l] = (float)tok;
    }
}

__global__ void feats_prep_kernel(const float* __restrict__ feats,
                                  const int* __restrict__ pos,
                                  bf16_t* __restrict__ feats_bf) {
    int i = blockIdx.x * blockDim.x + threadIdx.x;
    if (i >= B_ * R_ * F_) return;
    const int b = i / (R_ * F_);
    const int rem = i - b * (R_ * F_);
    feats_bf[i] = (bf16_t)feats[(size_t)pos[b] * R_ * F_ + rem];
}

// mean over R, written straight into the static middle slice of x1
__global__ void favg_kernel(const float* __restrict__ feats,
                            const int* __restrict__ pos,
                            bf16_t* __restrict__ x1) {
    int i = blockIdx.x * blockDim.x + threadIdx.x;
    if (i >= B_ * F_) return;
    const int b = i >> 11, f = i & (F_ - 1);
    const float* src = feats + (size_t)pos[b] * R_ * F_ + f;
    float acc = 0.f;
    for (int r = 0; r < R_; ++r) acc += src[(size_t)r * F_];
    x1[(size_t)b * (D_ + F_ + E_) + D_ + f] = (bf16_t)(acc * (1.f / R_));
}

__global__ void init_state_kernel(float* h1, float* c1, float* h2, float* c2,
                                  bf16_t* h1_bf, bf16_t* h2_bf) {
    int i = blockIdx.x * blockDim.x + threadIdx.x;
    if (i >= B_ * D_) return;
    h1[i] = 0.f; c1[i] = 0.f; h2[i] = 0.f; c2[i] = 0.f;
    h1_bf[i] = (bf16_t)0.f; h2_bf[i] = (bf16_t)0.f;
}

// dynamic parts of x1 = [h2 | favg | e_t]
__global__ void x1_step_kernel(const float* __restrict__ h2,
                               const float* __restrict__ emb,
                               const int* __restrict__ seqs_s, int t,
                               bf16_t* __restrict__ x1) {
    int i = blockIdx.x * blockDim.x + threadIdx.x;
    if (i >= B_ * 2048) return;
    const int b = i >> 11, j = i & 2047;
    bf16_t* row = x1 + (size_t)b * (D_ + F_ + E_);
    if (j < D_) {
        row[j] = (bf16_t)h2[b * D_ + j];
    } else {
        const int e = j - D_;
        const int tok = seqs_s[b * L_ + t];
        row[D_ + F_ + e] = (bf16_t)emb[(size_t)tok * E_ + e];
    }
}

__global__ void lstm_cell_kernel(const float* __restrict__ g,
                                 const float* __restrict__ c,
                                 float* __restrict__ nh, float* __restrict__ nc,
                                 bf16_t* __restrict__ nh_bf) {
    int i = blockIdx.x * blockDim.x + threadIdx.x;
    if (i >= B_ * D_) return;
    const int b = i / D_, d = i - b * D_;
    const float* gr = g + (size_t)b * 4 * D_;
    const float gi = gr[d], gf = gr[D_ + d], gg = gr[2 * D_ + d], go = gr[3 * D_ + d];
    const float si = 1.f / (1.f + expf(-gi));
    const float sf = 1.f / (1.f + expf(-gf));
    const float so = 1.f / (1.f + expf(-go));
    const float c2 = sf * c[i] + si * tanhf(gg);
    const float h2 = so * tanhf(c2);
    nc[i] = c2; nh[i] = h2; nh_bf[i] = (bf16_t)h2;
}

__global__ void carry_update_kernel(const float* __restrict__ nh,
                                    const float* __restrict__ nc,
                                    const int* __restrict__ dec_len, int t,
                                    float* __restrict__ h, float* __restrict__ c,
                                    bf16_t* __restrict__ h_bf) {
    int i = blockIdx.x * blockDim.x + threadIdx.x;
    if (i >= B_ * D_) return;
    const int b = i / D_;
    const bool m = t < dec_len[b];
    const float hv = m ? nh[i] : h[i];
    const float cv = m ? nc[i] : c[i];
    h[i] = hv; c[i] = cv; h_bf[i] = (bf16_t)hv;
}

// scores[b,r] = sum_a relu(d_att[b,a] + img_att[b,r,a]) * Wa[a] + ba
__global__ void score_kernel(const float* __restrict__ d_att,
                             const float* __restrict__ img_att,
                             const float* __restrict__ Wa,
                             const float* __restrict__ ba,
                             float* __restrict__ scores) {
    const int lane = threadIdx.x;
    const int id = blockIdx.x * blockDim.y + threadIdx.y;  // wave per (b,r)
    if (id >= B_ * R_) return;
    const int b = id / R_, r = id - b * R_;
    const float* da = d_att + (size_t)b * A_;
    const float* ia = img_att + ((size_t)b * R_ + r) * A_;
    float acc = 0.f;
    for (int a = lane; a < A_; a += 32) {
        float v = da[a] + ia[a];
        v = v > 0.f ? v : 0.f;
        acc += v * Wa[a];
    }
#pragma unroll
    for (int m = 16; m > 0; m >>= 1) acc += __shfl_xor(acc, m, 32);
    if (lane == 0) scores[id] = acc + ba[0];
}

__global__ void softmax_kernel(const float* __restrict__ scores,
                               float* __restrict__ sig,
                               float* __restrict__ out_sig) {
    const int b = threadIdx.x;
    if (b >= B_) return;
    const float* s = scores + b * R_;
    float mx = -3.0e38f;
    for (int r = 0; r < R_; ++r) mx = fmaxf(mx, s[r]);
    float sum = 0.f;
    for (int r = 0; r < R_; ++r) sum += expf(s[r] - mx);
    const float inv = 1.f / sum;
    for (int r = 0; r < R_; ++r) {
        const float v = expf(s[r] - mx) * inv;
        sig[b * R_ + r] = v;
        if (out_sig) out_sig[b * R_ + r] = v;
    }
}

// x2 = [ attention-weighted feats | nh1 ]
__global__ void x2_kernel(const float* __restrict__ feats,
                          const int* __restrict__ pos,
                          const float* __restrict__ sig,
                          const bf16_t* __restrict__ nh1_bf,
                          bf16_t* __restrict__ x2) {
    int i = blockIdx.x * blockDim.x + threadIdx.x;
    if (i >= B_ * (F_ + D_)) return;
    const int b = i / (F_ + D_), j = i - b * (F_ + D_);
    if (j < F_) {
        const float* src = feats + (size_t)pos[b] * R_ * F_ + j;
        const float* sg = sig + b * R_;
        float acc = 0.f;
        for (int r = 0; r < R_; ++r) acc += sg[r] * src[(size_t)r * F_];
        x2[(size_t)b * (F_ + D_) + j] = (bf16_t)acc;
    } else {
        x2[(size_t)b * (F_ + D_) + j] = nh1_bf[b * D_ + (j - F_)];
    }
}

// ================= host side =================
static inline char* carve(char*& p, size_t bytes) {
    char* r = p;
    p += (bytes + 255) & ~(size_t)255;
    return r;
}

extern "C" void kernel_launch(void* const* d_in, const int* in_sizes, int n_in,
                              void* d_out, int out_size, void* d_ws, size_t ws_size,
                              hipStream_t stream) {
    (void)in_sizes; (void)n_in; (void)out_size; (void)ws_size;
    const float* feats = (const float*)d_in[0];
    const int*   sequences = (const int*)d_in[1];
    const int*   sizes = (const int*)d_in[2];
    const float* emb = (const float*)d_in[3];
    const float* W_ih1 = (const float*)d_in[4];
    const float* W_hh1 = (const float*)d_in[5];
    const float* b_ih1 = (const float*)d_in[6];
    const float* b_hh1 = (const float*)d_in[7];
    const float* Wf = (const float*)d_in[8];
    const float* bf = (const float*)d_in[9];
    const float* Wd = (const float*)d_in[10];
    const float* bd = (const float*)d_in[11];
    const float* Wa = (const float*)d_in[12];
    const float* ba = (const float*)d_in[13];
    const float* W_ih2 = (const float*)d_in[14];
    const float* W_hh2 = (const float*)d_in[15];
    const float* b_ih2 = (const float*)d_in[16];
    const float* b_hh2 = (const float*)d_in[17];
    const float* Wo = (const float*)d_in[18];
    const float* bo = (const float*)d_in[19];

    float* out = (float*)d_out;
    float* out_preds = out;                               // (B, T, V)
    float* out_seqs  = out_preds + (size_t)B_ * T_ * V_;  // (B, L)
    float* out_dlen  = out_seqs + B_ * L_;                // (B,)
    float* out_sig   = out_dlen + B_;                     // (B, R)
    float* out_pos   = out_sig + B_ * R_;                 // (B,)

    char* p = (char*)d_ws;
    bf16_t* Wih1_bf = (bf16_t*)carve(p, (size_t)4 * D_ * (E_ + F_ + D_) * 2);
    bf16_t* Whh1_bf = (bf16_t*)carve(p, (size_t)4 * D_ * D_ * 2);
    bf16_t* Wih2_bf = (bf16_t*)carve(p, (size_t)4 * D_ * (F_ + D_) * 2);
    bf16_t* Whh2_bf = (bf16_t*)carve(p, (size_t)4 * D_ * D_ * 2);
    bf16_t* Wo_bf   = (bf16_t*)carve(p, (size_t)V_ * D_ * 2);
    bf16_t* Wf_bf   = (bf16_t*)carve(p, (size_t)A_ * F_ * 2);
    bf16_t* Wd_bf   = (bf16_t*)carve(p, (size_t)A_ * D_ * 2);
    bf16_t* feats_bf = (bf16_t*)carve(p, (size_t)B_ * R_ * F_ * 2);
    float*  img_att  = (float*)carve(p, (size_t)B_ * R_ * A_ * 4);
    bf16_t* x1 = (bf16_t*)carve(p, (size_t)B_ * (D_ + F_ + E_) * 2);
    bf16_t* x2 = (bf16_t*)carve(p, (size_t)B_ * (F_ + D_) * 2);
    float* g1 = (float*)carve(p, (size_t)B_ * 4 * D_ * 4);
    float* g2 = (float*)carve(p, (size_t)B_ * 4 * D_ * 4);
    float* h1 = (float*)carve(p, (size_t)B_ * D_ * 4);
    float* c1 = (float*)carve(p, (size_t)B_ * D_ * 4);
    float* h2 = (float*)carve(p, (size_t)B_ * D_ * 4);
    float* c2 = (float*)carve(p, (size_t)B_ * D_ * 4);
    float* nh1 = (float*)carve(p, (size_t)B_ * D_ * 4);
    float* nc1 = (float*)carve(p, (size_t)B_ * D_ * 4);
    float* nh2 = (float*)carve(p, (size_t)B_ * D_ * 4);
    float* nc2 = (float*)carve(p, (size_t)B_ * D_ * 4);
    bf16_t* h1_bf  = (bf16_t*)carve(p, (size_t)B_ * D_ * 2);
    bf16_t* h2_bf  = (bf16_t*)carve(p, (size_t)B_ * D_ * 2);
    bf16_t* nh1_bf = (bf16_t*)carve(p, (size_t)B_ * D_ * 2);
    bf16_t* nh2_bf = (bf16_t*)carve(p, (size_t)B_ * D_ * 2);
    float* d_att  = (float*)carve(p, (size_t)B_ * A_ * 4);
    float* scores = (float*)carve(p, (size_t)B_ * R_ * 4);
    float* sig    = (float*)carve(p, (size_t)B_ * R_ * 4);
    int* pos_i  = (int*)carve(p, B_ * 4);
    int* dlen_i = (int*)carve(p, B_ * 4);
    int* seqs_s = (int*)carve(p, B_ * L_ * 4);

    const dim3 wblk(32, 8);  // 8 waves; each wave: one 64x16 strip
    auto eb = [](int n) { return dim3((n + 255) / 256); };

    // ---- one-time prep ----
    sort_kernel<<<1, 64, 0, stream>>>(sizes, sequences, pos_i, dlen_i, seqs_s,
                                      out_seqs, out_dlen, out_pos);
    cvt_f32_bf16_kernel<<<eb(4 * D_ * (E_ + F_ + D_)), 256, 0, stream>>>(W_ih1, Wih1_bf, 4 * D_ * (E_ + F_ + D_));
    cvt_f32_bf16_kernel<<<eb(4 * D_ * D_), 256, 0, stream>>>(W_hh1, Whh1_bf, 4 * D_ * D_);
    cvt_f32_bf16_kernel<<<eb(4 * D_ * (F_ + D_)), 256, 0, stream>>>(W_ih2, Wih2_bf, 4 * D_ * (F_ + D_));
    cvt_f32_bf16_kernel<<<eb(4 * D_ * D_), 256, 0, stream>>>(W_hh2, Whh2_bf, 4 * D_ * D_);
    cvt_f32_bf16_kernel<<<eb(V_ * D_), 256, 0, stream>>>(Wo, Wo_bf, V_ * D_);
    cvt_f32_bf16_kernel<<<eb(A_ * F_), 256, 0, stream>>>(Wf, Wf_bf, A_ * F_);
    cvt_f32_bf16_kernel<<<eb(A_ * D_), 256, 0, stream>>>(Wd, Wd_bf, A_ * D_);
    feats_prep_kernel<<<eb(B_ * R_ * F_), 256, 0, stream>>>(feats, pos_i, feats_bf);
    favg_kernel<<<eb(B_ * F_), 256, 0, stream>>>(feats, pos_i, x1);
    init_state_kernel<<<eb(B_ * D_), 256, 0, stream>>>(h1, c1, h2, c2, h1_bf, h2_bf);

    // img_att = feats_s @ Wf^T + bf : M=2304 (36 strips), N=512, K=2048
    gemm_kernel<<<dim3(A_ / 128, (B_ * R_) / 64), wblk, 0, stream>>>(
        feats_bf, F_, Wf_bf, F_, F_, bf, img_att, A_);

    // ---- decode steps ----
    for (int t = 0; t < T_; ++t) {
        x1_step_kernel<<<eb(B_ * 2048), 256, 0, stream>>>(h2, emb, seqs_s, t, x1);

        // g1 = x1 @ W_ih1^T + h1 @ W_hh1^T + biases  (N = 4096)
        gemm_dual_kernel<<<dim3((4 * D_) / 128, 1), wblk, 0, stream>>>(
            x1, D_ + F_ + E_, D_ + F_ + E_, Wih1_bf, D_ + F_ + E_,
            h1_bf, D_, D_, Whh1_bf, D_,
            b_ih1, b_hh1, g1, 4 * D_);
        lstm_cell_kernel<<<eb(B_ * D_), 256, 0, stream>>>(g1, c1, nh1, nc1, nh1_bf);
        carry_update_kernel<<<eb(B_ * D_), 256, 0, stream>>>(nh1, nc1, dlen_i, t, h1, c1, h1_bf);

        // attention: d_att = nh1 @ Wd^T + bd  (M=64, N=512)
        gemm_kernel<<<dim3(A_ / 128, 1), wblk, 0, stream>>>(
            nh1_bf, D_, Wd_bf, D_, D_, bd, d_att, A_);
        score_kernel<<<dim3((B_ * R_ + 7) / 8), wblk, 0, stream>>>(d_att, img_att, Wa, ba, scores);
        softmax_kernel<<<1, 64, 0, stream>>>(scores, sig, (t == T_ - 1) ? out_sig : (float*)nullptr);
        x2_kernel<<<eb(B_ * (F_ + D_)), 256, 0, stream>>>(feats, pos_i, sig, nh1_bf, x2);

        // g2 = x2 @ W_ih2^T + h2 @ W_hh2^T + biases
        gemm_dual_kernel<<<dim3((4 * D_) / 128, 1), wblk, 0, stream>>>(
            x2, F_ + D_, F_ + D_, Wih2_bf, F_ + D_,
            h2_bf, D_, D_, Whh2_bf, D_,
            b_ih2, b_hh2, g2, 4 * D_);
        lstm_cell_kernel<<<eb(B_ * D_), 256, 0, stream>>>(g2, c2, nh2, nc2, nh2_bf);

        // masked logits straight into d_out: N=32000 -> 250 blocks
        logits_kernel<<<dim3(V_ / 128, 1), wblk, 0, stream>>>(
            nh2_bf, Wo_bf, bo, dlen_i, t, out_preds);

        carry_update_kernel<<<eb(B_ * D_), 256, 0, stream>>>(nh2, nc2, dlen_i, t, h2, c2, h2_bf);
    }
}
